// Encoder_78855599555052
// MI455X (gfx1250) — compile-verified, matrix operand order checked by reading, not run on hardware
//
#include <hip/hip_runtime.h>

// Problem constants (match reference)
#define N_NODES 30000
#define N_EDGES 480000
#define EF_EDGES (N_EDGES + N_NODES)
#define DIM     128
#define HEADS   8
#define CHD     16
#define EDIMC   16
#define NLAYERS 3
#define NGRAPH  256
#define DFFC    512

typedef __attribute__((ext_vector_type(16))) __bf16          v16bf;
typedef __attribute__((ext_vector_type(8)))  float           v8f;
typedef __attribute__((ext_vector_type(8)))  unsigned short  us8;
typedef unsigned short u16;
typedef long long ll;

// ---------------------------------------------------------------- helpers
__device__ __forceinline__ u16 f2bfu(float f) {
  unsigned u = __float_as_uint(f);
  u += 0x7fffu + ((u >> 16) & 1u);          // round-to-nearest-even
  return (u16)(u >> 16);
}

__device__ __forceinline__ void atomicMaxF(float* addr, float val) {
  unsigned* ua = (unsigned*)addr;
  unsigned old = *ua;
  while (__uint_as_float(old) < val) {
    unsigned assumed = old;
    old = atomicCAS(ua, assumed, __float_as_uint(val));
    if (old == assumed) break;
  }
}

// CDNA5 async memory->LDS DMA (VGLOBAL op 98, tracked by ASYNCcnt).
// lds_off: 32-bit LDS byte offset (low 32 bits of generic shared address,
// as mapped by the shared aperture); gaddr: 64-bit global address.
__device__ __forceinline__ void async_ld_b128(unsigned lds_off, const void* gaddr) {
  asm volatile("global_load_async_to_lds_b128 %0, %1, off"
               :: "v"(lds_off), "v"(gaddr) : "memory");
}
__device__ __forceinline__ void wait_async0() {
  asm volatile("s_wait_asynccnt 0x0" ::: "memory");
}
// Compiler scheduling fence: LDS fragment loads may not sink past this,
// so the following WMMA chain issues back-to-back (ISA 7.5 XDL note).
__device__ __forceinline__ void sched_fence() {
  asm volatile("" ::: "memory");
}

// ---------------------------------------------------------------- tiny utility kernels
__global__ void k_fill(float* p, float v, ll n) {
  ll g = (ll)blockIdx.x * blockDim.x + threadIdx.x;
  if (g < n) p[g] = v;
}

__global__ void k_copy(const float* __restrict__ s, float* __restrict__ d, ll n) {
  ll g = (ll)blockIdx.x * blockDim.x + threadIdx.x;
  if (g < n) d[g] = s[g];
}

__global__ void k_f2bf(const float* __restrict__ s, u16* __restrict__ d, ll n) {
  ll g = (ll)blockIdx.x * blockDim.x + threadIdx.x;
  if (g < n) d[g] = f2bfu(s[g]);
}

__global__ void k_rowbias(float* __restrict__ out, const float* __restrict__ b, ll n) {
  ll g = (ll)blockIdx.x * blockDim.x + threadIdx.x;
  if (g < n) out[g] = b[g & (DIM - 1)];
}

// ---------------------------------------------------------------- WMMA GEMM (bf16 in, f32 out)
// C[M,Ncols] = act(A[M,K] @ B[K,Ncols] + bias)   A,B bf16 (as u16), row-major.
// block = 256 threads (8 waves), tile = 128x128 out, K tiled by 128.
// A tile: async DMA global->LDS.  B tile: transpose-on-store into [col][k].
__global__ __launch_bounds__(256)
void gemm_wmma_bf16(const u16* __restrict__ A, const u16* __restrict__ B,
                    const float* __restrict__ bias, float* __restrict__ Cmat,
                    int M, int Ncols, int K, int act)
{
  __shared__ __align__(16) u16 As[128][128];   // [row][k]
  __shared__ __align__(16) u16 Bs[128][128];   // [col][k]  (transposed)

  const int t    = threadIdx.x;
  const int lane = t & 31;
  const int wave = t >> 5;
  const int hlf  = lane >> 4;       // which 16-lane half of wave32
  const int l16  = lane & 15;
  const int rowBlock = blockIdx.x * 128;
  const int colBlock = blockIdx.y * 128;

  v8f acc[8] = {};
  union Frag { v16bf v; us8 u[2]; };

  // pre-zero A-tile rows beyond M once (their async loads are EXEC-masked off;
  // these rows feed only output rows that are never stored)
  #pragma unroll
  for (int i = 0; i < 8; ++i) {
    int linear = (t + i * 256) * 8;
    int r = linear >> 7;
    int k = linear & 127;
    if (rowBlock + r >= M) {
      us8 z;
      #pragma unroll
      for (int j = 0; j < 8; ++j) z[j] = 0;
      *(us8*)&As[r][k] = z;
    }
  }

  for (int kt = 0; kt < K; kt += 128) {
    // --- A tile: async DMA straight into LDS (no VGPR bounce) ---
    #pragma unroll
    for (int i = 0; i < 8; ++i) {
      int linear = (t + i * 256) * 8;       // element index in 128x128 tile
      int r = linear >> 7;
      int k = linear & 127;
      int gr = rowBlock + r;
      if (gr < M) {
        unsigned loff = (unsigned)(size_t)&As[r][k];
        async_ld_b128(loff, (const void*)(A + (size_t)gr * K + kt + k));
      }
    }
    // --- B tile: vector load, transpose into [col][k] ---
    #pragma unroll
    for (int i = 0; i < 8; ++i) {
      int linear = (t + i * 256) * 8;
      int kk = linear >> 7;
      int c0 = linear & 127;
      us8 v = *(const us8*)(B + (size_t)(kt + kk) * Ncols + colBlock + c0);
      #pragma unroll
      for (int j = 0; j < 8; ++j) Bs[c0 + j][kk] = v[j];
    }
    wait_async0();
    __syncthreads();

    // --- 4 k-steps of 32: preload all fragments, then 8 back-to-back WMMAs ---
    #pragma unroll
    for (int ks = 0; ks < 4; ++ks) {
      const int k0  = ks * 32;
      const int klo = k0 + hlf * 8;         // K {0..7 | 8..15} halves
      const int khi = k0 + 16 + hlf * 8;    // K {16..23 | 24..31}
      const int arow = wave * 16 + l16;
      Frag fa;
      fa.u[0] = *(const us8*)&As[arow][klo];
      fa.u[1] = *(const us8*)&As[arow][khi];
      Frag fb[8];
      #pragma unroll
      for (int nt = 0; nt < 8; ++nt) {
        const int bcol = nt * 16 + l16;
        fb[nt].u[0] = *(const us8*)&Bs[bcol][klo];
        fb[nt].u[1] = *(const us8*)&Bs[bcol][khi];
      }
      sched_fence();   // keep the 18 ds_loads above; fire WMMAs back-to-back
      #pragma unroll
      for (int nt = 0; nt < 8; ++nt) {
        acc[nt] = __builtin_amdgcn_wmma_f32_16x16x32_bf16(
            false, fa.v, false, fb[nt].v, (short)0, acc[nt], false, false);
      }
    }
    __syncthreads();
  }

  // --- epilogue: bias + activation + store ---
  #pragma unroll
  for (int nt = 0; nt < 8; ++nt) {
    const int col = colBlock + nt * 16 + l16;
    const float bv = bias ? bias[col] : 0.0f;
    #pragma unroll
    for (int j = 0; j < 8; ++j) {
      const int r = rowBlock + wave * 16 + hlf * 8 + j;   // C/D layout: VGPR j -> M=j (+8 for hi half)
      if (r < M) {
        float v = acc[nt][j] + bv;
        if (act == 1) v = v > 0.0f ? v : 0.0f;            // ReLU
        Cmat[(size_t)r * Ncols + col] = v;
      }
    }
  }
}

// ---------------------------------------------------------------- GAT pieces
// Wae[k][h] = sum_c We[k, h*16+c] * a_edge[h, c]   (128 threads)
__global__ void k_wae(const float* __restrict__ We, const float* __restrict__ ae,
                      float* __restrict__ wae) {
  int g = threadIdx.x;           // 0..127
  int k = g >> 3, h = g & 7;
  float s = 0.f;
  #pragma unroll
  for (int c = 0; c < CHD; ++c) s += We[k * DIM + h * CHD + c] * ae[h * CHD + c];
  wae[k * HEADS + h] = s;
}

// a_s[n,h], a_d[n,h] from x
__global__ void k_asad(const float* __restrict__ x, const float* __restrict__ asrc,
                       const float* __restrict__ adst, float* __restrict__ a_s,
                       float* __restrict__ a_d, int n) {
  ll g = (ll)blockIdx.x * blockDim.x + threadIdx.x;
  if (g >= (ll)n * HEADS) return;
  int nn = (int)(g >> 3), h = (int)(g & 7);
  const float* xr = x + (size_t)nn * DIM + h * CHD;
  float s = 0.f, d = 0.f;
  #pragma unroll
  for (int c = 0; c < CHD; ++c) {
    s += xr[c] * asrc[h * CHD + c];
    d += xr[c] * adst[h * CHD + c];
  }
  a_s[g] = s; a_d[g] = d;
}

// self-loop attr accumulation: loop_attr[dst] += ew[e]; cnt[dst] += 1
__global__ void k_loop_accum(const int* __restrict__ ei, const float* __restrict__ ew,
                             float* __restrict__ loop_attr, float* __restrict__ cnt) {
  ll g = (ll)blockIdx.x * blockDim.x + threadIdx.x;
  if (g >= (ll)N_EDGES * EDIMC) return;
  int e = (int)(g >> 4), j = (int)(g & 15);
  int dd = ei[N_EDGES + e];
  atomicAdd(&loop_attr[(size_t)dd * EDIMC + j], ew[(size_t)e * EDIMC + j]);
  if (j == 0) atomicAdd(&cnt[dd], 1.0f);
}

__global__ void k_loop_div(float* __restrict__ loop_attr, const float* __restrict__ cnt, int n) {
  ll g = (ll)blockIdx.x * blockDim.x + threadIdx.x;
  if (g >= (ll)n * EDIMC) return;
  loop_attr[g] /= fmaxf(cnt[g >> 4], 1.0f);
}

// attention logits + segment max
__global__ void k_alpha(const int* __restrict__ ei, const float* __restrict__ ew,
                        const float* __restrict__ loop_attr, const float* __restrict__ a_s,
                        const float* __restrict__ a_d, const float* __restrict__ wae,
                        float* __restrict__ alpha, float* __restrict__ m) {
  ll g = (ll)blockIdx.x * blockDim.x + threadIdx.x;
  if (g >= (ll)EF_EDGES * HEADS) return;
  int e = (int)(g >> 3), h = (int)(g & 7);
  int s, dd; const float* ewp;
  if (e < N_EDGES) { s = ei[e]; dd = ei[N_EDGES + e]; ewp = ew + (size_t)e * EDIMC; }
  else             { s = dd = e - N_EDGES;            ewp = loop_attr + (size_t)(e - N_EDGES) * EDIMC; }
  float a = a_s[(size_t)s * HEADS + h] + a_d[(size_t)dd * HEADS + h];
  #pragma unroll
  for (int k = 0; k < EDIMC; ++k) a += ewp[k] * wae[k * HEADS + h];
  a = a > 0.f ? a : 0.2f * a;                // GAT leaky relu
  alpha[g] = a;
  atomicMaxF(&m[(size_t)dd * HEADS + h], a);
}

// exp + segment-sum denom
__global__ void k_exp(const int* __restrict__ ei, const float* __restrict__ m,
                      float* __restrict__ alpha, float* __restrict__ denom) {
  ll g = (ll)blockIdx.x * blockDim.x + threadIdx.x;
  if (g >= (ll)EF_EDGES * HEADS) return;
  int e = (int)(g >> 3), h = (int)(g & 7);
  int dd = (e < N_EDGES) ? ei[N_EDGES + e] : e - N_EDGES;
  float p = __expf(alpha[g] - m[(size_t)dd * HEADS + h]);
  alpha[g] = p;
  atomicAdd(&denom[(size_t)dd * HEADS + h], p);
}

// weighted scatter aggregation: out[dst, d] += (alpha/denom) * x[src, d]
__global__ void k_aggregate(const int* __restrict__ ei, const float* __restrict__ alpha,
                            const float* __restrict__ denom, const float* __restrict__ x,
                            float* __restrict__ out) {
  ll g = (ll)blockIdx.x * blockDim.x + threadIdx.x;
  if (g >= (ll)EF_EDGES * DIM) return;
  int e = (int)(g >> 7), d = (int)(g & 127), h = d >> 4;
  int s, dd;
  if (e < N_EDGES) { s = ei[e]; dd = ei[N_EDGES + e]; }
  else             { s = dd = e - N_EDGES; }
  float w = alpha[(size_t)e * HEADS + h] / denom[(size_t)dd * HEADS + h];
  atomicAdd(&out[(size_t)dd * DIM + d], w * x[(size_t)s * DIM + d]);
}

// LayerNorm (+ optional leaky + residual). One wave32 per row of 128.
__global__ void k_ln(const float* __restrict__ x, const float* __restrict__ gw,
                     const float* __restrict__ bw, float* __restrict__ dst,
                     int n, float slope, int residual) {
  int row  = blockIdx.x * 8 + (threadIdx.x >> 5);
  int lane = threadIdx.x & 31;
  if (row >= n) return;
  const float* xr = x + (size_t)row * DIM;
  float v[4], s = 0.f, s2 = 0.f;
  #pragma unroll
  for (int i = 0; i < 4; ++i) { v[i] = xr[lane + 32 * i]; s += v[i]; s2 += v[i] * v[i]; }
  #pragma unroll
  for (int msk = 16; msk > 0; msk >>= 1) {
    s  += __shfl_xor(s,  msk, 32);
    s2 += __shfl_xor(s2, msk, 32);
  }
  float mu  = s / 128.f;
  float var = s2 / 128.f - mu * mu;
  float inv = rsqrtf(var + 1e-6f);
  #pragma unroll
  for (int i = 0; i < 4; ++i) {
    int d = lane + 32 * i;
    float y = (v[i] - mu) * inv * gw[d] + bw[d];
    if (residual) {
      y = y > 0.f ? y : slope * y;            // leaky relu
      dst[(size_t)row * DIM + d] += y;
    } else {
      dst[(size_t)row * DIM + d] = y;
    }
  }
}

// pooled segment sum into d_out
__global__ void k_pool(const float* __restrict__ x, const int* __restrict__ batch,
                       float* __restrict__ out) {
  ll g = (ll)blockIdx.x * blockDim.x + threadIdx.x;
  if (g >= (ll)N_NODES * DIM) return;
  int nn = (int)(g >> 7), d = (int)(g & 127);
  atomicAdd(&out[(size_t)batch[nn] * DIM + d], x[g]);
}

// ---------------------------------------------------------------- host side
static inline unsigned nblk(ll n) { return (unsigned)((n + 255) / 256); }

extern "C" void kernel_launch(void* const* d_in, const int* in_sizes, int n_in,
                              void* d_out, int out_size, void* d_ws, size_t ws_size,
                              hipStream_t stream) {
  (void)in_sizes; (void)n_in; (void)out_size; (void)ws_size;
  const float* nf_in = (const float*)d_in[0];
  const int*   ei    = (const int*)  d_in[1];   // [2,E]
  const float* ew    = (const float*)d_in[2];   // [E,16]
  const int*   batch = (const int*)  d_in[3];   // [N]

  struct GatP { const float *W, *as_, *ad_, *We, *ae_, *b; };
  GatP gp[2];
  gp[0] = { (const float*)d_in[4],  (const float*)d_in[5],  (const float*)d_in[6],
            (const float*)d_in[7],  (const float*)d_in[8],  (const float*)d_in[9]  };
  gp[1] = { (const float*)d_in[10], (const float*)d_in[11], (const float*)d_in[12],
            (const float*)d_in[13], (const float*)d_in[14], (const float*)d_in[15] };
  const float* lng[3] = { (const float*)d_in[16], (const float*)d_in[18], (const float*)d_in[20] };
  const float* lnb[3] = { (const float*)d_in[17], (const float*)d_in[19], (const float*)d_in[21] };
  const float* Wff1 = (const float*)d_in[22];
  const float* bff1 = (const float*)d_in[23];
  const float* Wff2 = (const float*)d_in[24];
  const float* bff2 = (const float*)d_in[25];
  const float* gF   = (const float*)d_in[26];
  const float* bF   = (const float*)d_in[27];

  // ---- workspace partition ----
  char* wsp = (char*)d_ws;
  auto wal = [&](size_t bytes) -> char* {
    char* p = wsp; wsp += (bytes + 255) & ~(size_t)255; return p;
  };
  float* nf   = (float*)wal(sizeof(float) * (size_t)N_NODES * DIM);
  u16*   abf  = (u16*)  wal(sizeof(u16)   * (size_t)N_NODES * DFFC);   // bf16 activations (reused)
  float* xbuf = (float*)wal(sizeof(float) * (size_t)N_NODES * DIM);
  float* gout = (float*)wal(sizeof(float) * (size_t)N_NODES * DIM);
  float* h1   = (float*)wal(sizeof(float) * (size_t)N_NODES * DFFC);
  float* a_s  = (float*)wal(sizeof(float) * (size_t)N_NODES * HEADS);
  float* a_d  = (float*)wal(sizeof(float) * (size_t)N_NODES * HEADS);
  float* mbuf = (float*)wal(sizeof(float) * (size_t)N_NODES * HEADS);
  float* dbuf = (float*)wal(sizeof(float) * (size_t)N_NODES * HEADS);
  float* alp  = (float*)wal(sizeof(float) * (size_t)EF_EDGES * HEADS);
  float* lat  = (float*)wal(sizeof(float) * (size_t)N_NODES * EDIMC);
  float* cnt  = (float*)wal(sizeof(float) * (size_t)N_NODES);
  float* wae  = (float*)wal(sizeof(float) * EDIMC * HEADS);
  u16*   W1b  = (u16*)  wal(sizeof(u16) * (size_t)NLAYERS * DIM * DIM);
  u16*   W2b  = (u16*)  wal(sizeof(u16) * (size_t)NLAYERS * DIM * DIM);
  u16*   F1b  = (u16*)  wal(sizeof(u16) * (size_t)NLAYERS * DIM * DFFC);
  u16*   F2b  = (u16*)  wal(sizeof(u16) * (size_t)NLAYERS * DFFC * DIM);

  // ---- weights -> bf16 ----
  k_f2bf<<<nblk((ll)NLAYERS*DIM*DIM),  256, 0, stream>>>(gp[0].W, W1b, (ll)NLAYERS*DIM*DIM);
  k_f2bf<<<nblk((ll)NLAYERS*DIM*DIM),  256, 0, stream>>>(gp[1].W, W2b, (ll)NLAYERS*DIM*DIM);
  k_f2bf<<<nblk((ll)NLAYERS*DIM*DFFC), 256, 0, stream>>>(Wff1, F1b, (ll)NLAYERS*DIM*DFFC);
  k_f2bf<<<nblk((ll)NLAYERS*DFFC*DIM), 256, 0, stream>>>(Wff2, F2b, (ll)NLAYERS*DFFC*DIM);

  // ---- self-loop attrs (mean of incoming edge attrs) ----
  k_fill<<<nblk((ll)N_NODES*EDIMC), 256, 0, stream>>>(lat, 0.f, (ll)N_NODES*EDIMC);
  k_fill<<<nblk((ll)N_NODES),       256, 0, stream>>>(cnt, 0.f, (ll)N_NODES);
  k_loop_accum<<<nblk((ll)N_EDGES*EDIMC), 256, 0, stream>>>(ei, ew, lat, cnt);
  k_loop_div<<<nblk((ll)N_NODES*EDIMC),   256, 0, stream>>>(lat, cnt, N_NODES);

  // ---- nf working copy ----
  k_copy<<<nblk((ll)N_NODES*DIM), 256, 0, stream>>>(nf_in, nf, (ll)N_NODES*DIM);

  const unsigned lnBlocks = (N_NODES + 7) / 8;

  for (int li = 0; li < NLAYERS; ++li) {
    for (int g = 0; g < 2; ++g) {
      const GatP& P = gp[g];
      const u16* Wb = (g == 0 ? W1b : W2b) + (size_t)li * DIM * DIM;
      // x = nf @ W[li]
      k_f2bf<<<nblk((ll)N_NODES*DIM), 256, 0, stream>>>(nf, abf, (ll)N_NODES*DIM);
      {
        dim3 grid((N_NODES + 127) / 128, DIM / 128);
        gemm_wmma_bf16<<<grid, 256, 0, stream>>>(abf, Wb, nullptr, xbuf,
                                                 N_NODES, DIM, DIM, 0);
      }
      k_wae<<<1, 128, 0, stream>>>(P.We + (size_t)li * EDIMC * DIM,
                                   P.ae_ + (size_t)li * HEADS * CHD, wae);
      k_asad<<<nblk((ll)N_NODES*HEADS), 256, 0, stream>>>(
          xbuf, P.as_ + (size_t)li * HEADS * CHD, P.ad_ + (size_t)li * HEADS * CHD,
          a_s, a_d, N_NODES);
      k_fill<<<nblk((ll)N_NODES*HEADS), 256, 0, stream>>>(mbuf, -1e30f, (ll)N_NODES*HEADS);
      k_fill<<<nblk((ll)N_NODES*HEADS), 256, 0, stream>>>(dbuf, 0.f,    (ll)N_NODES*HEADS);
      k_alpha<<<nblk((ll)EF_EDGES*HEADS), 256, 0, stream>>>(ei, ew, lat, a_s, a_d, wae, alp, mbuf);
      k_exp<<<nblk((ll)EF_EDGES*HEADS),   256, 0, stream>>>(ei, mbuf, alp, dbuf);
      k_rowbias<<<nblk((ll)N_NODES*DIM),  256, 0, stream>>>(gout, P.b + (size_t)li * DIM,
                                                            (ll)N_NODES*DIM);
      k_aggregate<<<nblk((ll)EF_EDGES*DIM), 256, 0, stream>>>(ei, alp, dbuf, xbuf, gout);
      k_ln<<<lnBlocks, 256, 0, stream>>>(gout, lng[g] + (size_t)li * DIM,
                                         lnb[g] + (size_t)li * DIM, nf, N_NODES, 0.01f, 1);
    }
    // ---- FFN ----
    k_f2bf<<<nblk((ll)N_NODES*DIM), 256, 0, stream>>>(nf, abf, (ll)N_NODES*DIM);
    {
      dim3 grid((N_NODES + 127) / 128, DFFC / 128);
      gemm_wmma_bf16<<<grid, 256, 0, stream>>>(abf, F1b + (size_t)li * DIM * DFFC,
                                               bff1 + (size_t)li * DFFC, h1,
                                               N_NODES, DFFC, DIM, 1 /*relu*/);
    }
    k_f2bf<<<nblk((ll)N_NODES*DFFC), 256, 0, stream>>>(h1, abf, (ll)N_NODES*DFFC);
    {
      dim3 grid((N_NODES + 127) / 128, DIM / 128);
      gemm_wmma_bf16<<<grid, 256, 0, stream>>>(abf, F2b + (size_t)li * DFFC * DIM,
                                               bff2 + (size_t)li * DIM, gout,
                                               N_NODES, DIM, DFFC, 0);
    }
    k_ln<<<lnBlocks, 256, 0, stream>>>(gout, lng[2] + (size_t)li * DIM,
                                       lnb[2] + (size_t)li * DIM, nf, N_NODES, 0.01f, 1);
  }

  // ---- final LN + pool ----
  k_ln<<<lnBlocks, 256, 0, stream>>>(nf, gF, bF, xbuf, N_NODES, 0.f, 0);
  k_fill<<<nblk((ll)NGRAPH*DIM), 256, 0, stream>>>((float*)d_out, 0.f, (ll)NGRAPH*DIM);
  k_pool<<<nblk((ll)N_NODES*DIM), 256, 0, stream>>>(xbuf, batch, (float*)d_out);
}
